// ScaledDotProductAttention_27659589386439
// MI455X (gfx1250) — matrix operand order repaired and checked by reading, not compiled, not run on hardware
//
#include <hip/hip_runtime.h>
#include <hip/hip_bf16.h>

typedef __attribute__((ext_vector_type(16))) _Float16 v16h;
typedef __attribute__((ext_vector_type(8)))  float    v8f;

#define S_LEN 2048
#define D_DIM 64
#define NHEAD 16
#define NBATCH 2
#define QT 16                 // query rows per workgroup
#define SS 2052               // LDS row stride (floats), +4 skew vs 2048
#define NEG_INF_F (-1.0e9f)

static __device__ __forceinline__ v8f wmma_f16(v16h a, v16h b, v8f c) {
  // D = A(16x32 f16) * B(32x16 f16) + C(16x16 f32)
  return __builtin_amdgcn_wmma_f32_16x16x32_f16(
      /*neg_a=*/false, a, /*neg_b=*/false, b,
      /*c_mod=*/(short)0, c, /*reuse_a=*/false, /*reuse_b=*/false);
}

__global__ __launch_bounds__(256)
void sdpa_fa_kernel(const float* __restrict__ Q,
                    const float* __restrict__ K,
                    const float* __restrict__ V,
                    const float* __restrict__ bias,
                    float* __restrict__ ctx_out,
                    float* __restrict__ attn_out) {
  extern __shared__ float smem[];
  float* sScores = smem;                 // 16 * SS
  float* sRinv   = smem + 16 * SS;       // 16
  float* sRed    = sRinv + 16;           // 4 waves * 32 lanes * 8 = 1024

  const int qt  = blockIdx.x;            // 0..127
  const int bh  = blockIdx.y;            // 0..31
  const int q0  = qt * QT;
  const int tid = threadIdx.x;
  const int lane = tid & 31;
  const int w    = tid >> 5;             // wave id 0..7

  const float* Qb = Q    + (size_t)bh * S_LEN * D_DIM;
  const float* Kb = K    + (size_t)bh * S_LEN * D_DIM;
  const float* Vb = V    + (size_t)bh * S_LEN * D_DIM;
  const float* Bi = bias + (size_t)bh * S_LEN * S_LEN;
  float* ctxb  = ctx_out  + (size_t)bh * S_LEN * D_DIM;
  float* attnb = attn_out + (size_t)bh * S_LEN * S_LEN;

  const int ncols     = (qt + 1) * QT;            // valid causal columns
  const int ncols_pad = (ncols + 63) & ~63;       // pad to multiple of 64 for PV

  // ---- Build Q A-fragments once (A layout: lane=row, K 16-granular per half-wave) ----
  const int mA    = lane & 15;
  const int koffA = (lane >> 4) * 16;             // half-wave K offset
  v16h aq0, aq1;
  {
    const float* qrow = Qb + (size_t)(q0 + mA) * D_DIM;
#pragma unroll
    for (int h = 0; h < 8; ++h) {
      aq0[h]     = (_Float16)qrow[koffA + h];          // K = koff+h      (cols 0..31)
      aq0[h + 8] = (_Float16)qrow[koffA + 8 + h];      // K = koff+8+h
      aq1[h]     = (_Float16)qrow[32 + koffA + h];     // cols 32..63
      aq1[h + 8] = (_Float16)qrow[40 + koffA + h];
    }
  }

  // ---- Phase 1: scores = QK^T * scale + bias, causal mask, into LDS ----
  const int koffB  = (lane >> 4) * 16;            // B layout: K 16-granular per half-wave
  const int nB     = lane & 15;                   // B/C/D column = lane
  const int mrBase = (lane >> 4) * 8;             // C/D row base per half-wave
  const float scale = 0.125f;                     // 1/sqrt(64)

  for (int kt = w; kt <= qt; kt += 8) {
    const int k0 = kt * QT;
    const float* krow = Kb + (size_t)(k0 + nB) * D_DIM;
    v16h bk0, bk1;
#pragma unroll
    for (int h = 0; h < 16; ++h) {
      bk0[h] = (_Float16)krow[koffB + h];          // B[kdim][n] = K[k0+n][kdim]
      bk1[h] = (_Float16)krow[32 + koffB + h];
    }
    v8f acc = {};
    acc = wmma_f16(aq0, bk0, acc);
    acc = wmma_f16(aq1, bk1, acc);

    const float* brow = Bi + (size_t)(q0 + mrBase) * S_LEN + k0 + nB;
    if (kt + 8 <= qt)
      __builtin_prefetch(Bi + (size_t)(q0 + mrBase) * S_LEN + k0 + 128 + nB, 0, 0);
#pragma unroll
    for (int r = 0; r < 8; ++r) {
      const int mm = r + mrBase;                  // tile row
      const int gq = q0 + mm, gk = k0 + nB;
      float s = acc[r] * scale + brow[(size_t)r * S_LEN];
      if (gk > gq) s = NEG_INF_F;                 // causal: replace, like jnp.where
      sScores[mm * SS + gk] = s;
    }
  }
  // zero-pad [ncols, ncols_pad) so PV K=32 chunks read zeros
  {
    const int pad = ncols_pad - ncols;
    for (int i = tid; i < QT * pad; i += 256) {
      const int rr = i / pad, cc = i % pad;
      sScores[rr * SS + ncols + cc] = 0.0f;
    }
  }
  __syncthreads();

  // ---- Phase 2: row softmax in LDS (store exp, keep 1/sum) ----
  for (int rr = w; rr < QT; rr += 8) {
    float* row = &sScores[rr * SS];
    float mx = -3.4e38f;
    for (int c = lane; c < ncols; c += 32) mx = fmaxf(mx, row[c]);
#pragma unroll
    for (int off = 16; off > 0; off >>= 1) mx = fmaxf(mx, __shfl_xor(mx, off, 32));
    float sum = 0.0f;
    for (int c = lane; c < ncols; c += 32) {
      const float e = __expf(row[c] - mx);
      row[c] = e;
      sum += e;
    }
#pragma unroll
    for (int off = 16; off > 0; off >>= 1) sum += __shfl_xor(sum, off, 32);
    if (lane == 0) sRinv[rr] = 1.0f / sum;
  }
  __syncthreads();

  // ---- Phase 4a: PV partial accumulation (split n-chunk x K-half over 8 waves) ----
  const int n0 = (w & 3) * 16;                    // output column chunk
  const int nk = ncols_pad >> 5;                  // K chunks of 32 (even count)
  v8f pacc = {};
  for (int kb32 = (w >> 2); kb32 < nk; kb32 += 2) {
    const int kb = kb32 * 32;
    v16h ap, bv;
    const float* prow = &sScores[mA * SS + kb];
#pragma unroll
    for (int h = 0; h < 8; ++h) {
      ap[h]     = (_Float16)prow[koffA + h];
      ap[h + 8] = (_Float16)prow[koffA + 8 + h];
    }
#pragma unroll
    for (int h = 0; h < 16; ++h)
      bv[h] = (_Float16)Vb[(size_t)(kb + koffB + h) * D_DIM + n0 + nB];
    pacc = wmma_f16(ap, bv, pacc);
  }
  if (w >= 4) {
    float* dst = &sRed[((w - 4) * 32 + lane) * 8];
#pragma unroll
    for (int r = 0; r < 8; ++r) dst[r] = pacc[r];
  }

  // ---- Phase 3: write normalized attention rows (bandwidth-critical) ----
  for (int rr = w; rr < QT; rr += 8) {
    const float rinv = sRinv[rr];
    float4* dst = (float4*)(attnb + (size_t)(q0 + rr) * S_LEN);
    const float* row = &sScores[rr * SS];
    for (int c4 = lane; c4 < S_LEN / 4; c4 += 32) {
      const int c = c4 * 4;
      float4 v;
      if (c < ncols) {
        const float4 e = *(const float4*)&row[c];
        v.x = e.x * rinv; v.y = e.y * rinv; v.z = e.z * rinv; v.w = e.w * rinv;
      } else {
        v.x = v.y = v.z = v.w = 0.0f;
      }
      dst[c4] = v;
    }
  }
  __syncthreads();

  // ---- Phase 4b: reduce PV partials, scale by 1/sum, write context ----
  if (w < 4) {
    const float* src = &sRed[(w * 32 + lane) * 8];
#pragma unroll
    for (int r = 0; r < 8; ++r) {
      const int mm = r + mrBase;
      const float v = (pacc[r] + src[r]) * sRinv[mm];
      ctxb[(size_t)(q0 + mm) * D_DIM + n0 + nB] = v;
    }
  }
}

extern "C" void kernel_launch(void* const* d_in, const int* in_sizes, int n_in,
                              void* d_out, int out_size, void* d_ws, size_t ws_size,
                              hipStream_t stream) {
  const float* Q    = (const float*)d_in[0];
  const float* K    = (const float*)d_in[1];
  const float* V    = (const float*)d_in[2];
  // d_in[3] = attn_mask (causal, recomputed in-kernel; ignored)
  const float* bias = (const float*)d_in[4];

  float* ctx  = (float*)d_out;                                     // [B,H,S,D]
  float* attn = (float*)d_out + (size_t)NBATCH * NHEAD * S_LEN * D_DIM; // [B,H,S,S]

  const size_t smem_bytes = (size_t)(16 * SS + 16 + 1024) * sizeof(float);
  hipFuncSetAttribute((const void*)sdpa_fa_kernel,
                      hipFuncAttributeMaxDynamicSharedMemorySize, (int)smem_bytes);

  dim3 grid(S_LEN / QT, NBATCH * NHEAD);   // (128, 32)
  dim3 block(256);
  sdpa_fa_kernel<<<grid, block, smem_bytes, stream>>>(Q, K, V, bias, ctx, attn);
}
